// Model_36180804502056
// MI455X (gfx1250) — matrix operand order repaired and checked by reading, not compiled, not run on hardware
//
#include <hip/hip_runtime.h>
#include <hip/hip_bf16.h>
#include <math.h>

// ---------------- dims ----------------
constexpr int cN   = 4096;
constexpr int cT   = 48;
constexpr int cF   = 76;
constexpr int cFP  = 96;    // F padded to 3*32
constexpr int cH   = 252;
constexpr int cG3  = 768;   // 3H=756 padded to 48*16
constexpr int cD   = 256;   // H + DEMO
constexpr int cK   = 10;
constexpr int cFFN = 128;

typedef __attribute__((ext_vector_type(16))) _Float16 v16h;
typedef __attribute__((ext_vector_type(8)))  _Float16 v8h;
typedef __attribute__((ext_vector_type(8)))  float    v8f;

// ---------------- fragment helpers (CDNA5 wave32 WMMA layouts) ----------------
__device__ __forceinline__ v16h frag_combine(v8h lo, v8h hi) {
  v16h f;
  f[0]=lo[0];  f[1]=lo[1];  f[2]=lo[2];  f[3]=lo[3];
  f[4]=lo[4];  f[5]=lo[5];  f[6]=lo[6];  f[7]=lo[7];
  f[8]=hi[0];  f[9]=hi[1];  f[10]=hi[2]; f[11]=hi[3];
  f[12]=hi[4]; f[13]=hi[5]; f[14]=hi[6]; f[15]=hi[7];
  return f;
}

// A 16x32 f16 frag from row-major [rows][ld]: lane m=L%16, K = (L/16)*16 + j
__device__ __forceinline__ v16h fragA_rm(const _Float16* base, int row0, int ld, int k0) {
  int lane = threadIdx.x & 31;
  const _Float16* p = base + (size_t)(row0 + (lane & 15)) * ld + k0 + ((lane >> 4) << 4);
  return frag_combine(*(const v8h*)p, *(const v8h*)(p + 8));
}

// B 32x16 f16 frag where B[k][n] = src[n][k] (row-major src): lane n=L%16, K = (L/16)*16 + j
__device__ __forceinline__ v16h fragB_rm(const _Float16* base, int n0, int ld, int k0) {
  int lane = threadIdx.x & 31;
  const _Float16* p = base + (size_t)(n0 + (lane & 15)) * ld + k0 + ((lane >> 4) << 4);
  return frag_combine(*(const v8h*)p, *(const v8h*)(p + 8));
}

// B frag from pre-packed layout: tile = 512 halves, lane-contiguous
__device__ __forceinline__ v16h fragB_pk(const _Float16* tile) {
  int lane = threadIdx.x & 31;
  const _Float16* p = tile + lane * 16;
  return frag_combine(*(const v8h*)p, *(const v8h*)(p + 8));
}

__device__ __forceinline__ v8f wmma16(v16h a, v16h b, v8f c) {
  return __builtin_amdgcn_wmma_f32_16x16x32_f16(false, a, false, b, (short)0, c, false, false);
}

__device__ __forceinline__ float sigm(float x) { return 1.0f / (1.0f + __expf(-x)); }

// ---------------- packing kernels ----------------
// pack f32 weight into WMMA-B fragment layout; trans=1: B[k][n]=W[n*Kr+k], else W[k*Nr+n]
__global__ void pack_w(const float* __restrict__ W, int Kr, int Nr, int trans,
                       _Float16* __restrict__ outp, int KT) {
  size_t i = (size_t)blockIdx.x * 256 + threadIdx.x;
  int j = i & 15; int lane = (int)((i >> 4) & 31); size_t tile = i >> 9;
  int kt = (int)(tile % KT); int nt = (int)(tile / KT);
  int k = kt * 32 + ((lane >> 4) << 4) + j;
  int n = nt * 16 + (lane & 15);
  float v = 0.0f;
  if (k < Kr && n < Nr) v = trans ? W[(size_t)n * Kr + k] : W[(size_t)k * Nr + n];
  outp[i] = (_Float16)v;
}

// pack f16 row-major [K][ldn] into B fragment layout (runtime matrices, e.g. XW)
__global__ void pack_b16(const _Float16* __restrict__ src, int ldn, int KT,
                         _Float16* __restrict__ outp) {
  size_t i = (size_t)blockIdx.x * 256 + threadIdx.x;
  int j = i & 15; int lane = (int)((i >> 4) & 31); size_t tile = i >> 9;
  int kt = (int)(tile % KT); int nt = (int)(tile / KT);
  int k = kt * 32 + ((lane >> 4) << 4) + j;
  int n = nt * 16 + (lane & 15);
  outp[i] = src[(size_t)k * ldn + n];
}

// ---------------- misc prep ----------------
__global__ void init_zero(float* fsum, float* kl, float* cl) {
  int t = threadIdx.x;
  if (t < cK) fsum[t] = 0.0f;
  if (t == cK) kl[0] = 0.0f;
  if (t == cK + 1) cl[0] = 0.0f;
}

__global__ void convert_x(const float* __restrict__ x, _Float16* __restrict__ xh) {
  size_t i = (size_t)blockIdx.x * 256 + threadIdx.x;   // over T*N*FP
  int f = (int)(i % cFP); size_t tn = i / cFP;
  int n = (int)(tn % cN); int t = (int)(tn / cN);
  float v = (f < cF) ? x[((size_t)n * cT + t) * cF + f] : 0.0f;
  xh[i] = (_Float16)v;
}

__global__ void fill_demo(const float* __restrict__ xd, float* __restrict__ xs32,
                          _Float16* __restrict__ xs16) {
  int n = blockIdx.x * 256 + threadIdx.x;
  if (n >= cN) return;
  for (int j = 0; j < 4; ++j) {
    float v = xd[n * 4 + j];
    xs32[(size_t)n * cD + cH + j] = v;
    xs16[(size_t)n * cD + cH + j] = (_Float16)v;
  }
}

__global__ void build_cscale(const float* __restrict__ wo, float* __restrict__ cs) {
  int c = threadIdx.x;                // 256
  cs[c] = wo[c >> 6] * 0.125f;        // wo[head] / sqrt(DK=64)
}

// ---------------- GRU: 16 rows per block, full 48-step recurrence ----------------
__global__ void __launch_bounds__(256)
gru_kernel(const _Float16* __restrict__ xh, const _Float16* __restrict__ WihP,
           const _Float16* __restrict__ WhhP, const float* __restrict__ b_ih,
           const float* __restrict__ b_hh, const float* __restrict__ h0,
           const int* __restrict__ slen, float* __restrict__ xs32,
           _Float16* __restrict__ xs16) {
  __shared__ __align__(16) _Float16 sX[16 * cFP];
  __shared__ __align__(16) _Float16 sH[16 * cD];
  __shared__ __align__(16) _Float16 sGI[16 * cG3];
  __shared__ __align__(16) _Float16 sGH[16 * cG3];
  __shared__ int sIdx[16];
  const int tid = threadIdx.x, lane = tid & 31, wv = tid >> 5;
  const int r0 = blockIdx.x * 16;

  if (tid < 16) {
    int l = slen[r0 + tid] - 1;
    sIdx[tid] = l < 0 ? 0 : (l > cT - 1 ? cT - 1 : l);
  }
  for (int e = tid; e < 16 * cD; e += 256) {
    int c = e & (cD - 1);
    sH[e] = (c < cH) ? (_Float16)h0[c] : (_Float16)0.0f;
  }

  for (int t = 0; t < cT; ++t) {
    __syncthreads();
    {
      const _Float16* src = xh + ((size_t)t * cN + r0) * cFP;
      for (int e = tid; e < 16 * cFP; e += 256) sX[e] = src[e];
      if (t + 1 < cT && tid < 48)
        __builtin_prefetch(xh + ((size_t)(t + 1) * cN + r0) * cFP + tid * 32, 0, 1);
    }
    __syncthreads();
    // 96 tile-GEMMs: 48 gi tiles (K=96 over Wih) + 48 gh tiles (K=256 over Whh)
    for (int i = 0; i < 12; ++i) {
      int id = wv + 8 * i;
      v8f acc = {};
      int cm = ((lane >> 4) << 3), cn = lane & 15;
      if (id < 48) {
        int nt = id;
        #pragma unroll
        for (int kt = 0; kt < 3; ++kt)
          acc = wmma16(fragA_rm(sX, 0, cFP, kt * 32),
                       fragB_pk(WihP + ((size_t)(nt * 3 + kt)) * 512), acc);
        #pragma unroll
        for (int r = 0; r < 8; ++r) sGI[(cm + r) * cG3 + nt * 16 + cn] = (_Float16)acc[r];
      } else {
        int nt = id - 48;
        #pragma unroll
        for (int kt = 0; kt < 8; ++kt)
          acc = wmma16(fragA_rm(sH, 0, cD, kt * 32),
                       fragB_pk(WhhP + ((size_t)(nt * 8 + kt)) * 512), acc);
        #pragma unroll
        for (int r = 0; r < 8; ++r) sGH[(cm + r) * cG3 + nt * 16 + cn] = (_Float16)acc[r];
      }
    }
    __syncthreads();
    // gate math (torch GRU equations)
    for (int e = tid; e < 16 * cH; e += 256) {
      int m = e / cH, c = e - m * cH;
      float gir = (float)sGI[m * cG3 + c],          ghr = (float)sGH[m * cG3 + c];
      float giz = (float)sGI[m * cG3 + cH + c],     ghz = (float)sGH[m * cG3 + cH + c];
      float gin = (float)sGI[m * cG3 + 2 * cH + c], ghn = (float)sGH[m * cG3 + 2 * cH + c];
      float r  = sigm(gir + ghr + b_ih[c] + b_hh[c]);
      float zg = sigm(giz + ghz + b_ih[cH + c] + b_hh[cH + c]);
      float ng = tanhf(gin + b_ih[2 * cH + c] + r * (ghn + b_hh[2 * cH + c]));
      float hold = (float)sH[m * cD + c];
      float hn = (1.0f - zg) * ng + zg * hold;
      sH[m * cD + c] = (_Float16)hn;
      if (t == sIdx[m]) {
        size_t o = (size_t)(r0 + m) * cD + c;
        xs32[o] = hn; xs16[o] = (_Float16)hn;
      }
    }
  }
}

// ---------------- generic GEMM: C[M,N] = A[M,K]@Bpacked + bias, optional col scale --------
__global__ void __launch_bounds__(256)
gemm16(const _Float16* __restrict__ A, int lda, int KT,
       const _Float16* __restrict__ Bp, const float* __restrict__ bias,
       const float* __restrict__ cscale, _Float16* __restrict__ out16,
       float* __restrict__ out32, int ldo) {
  __shared__ __align__(16) _Float16 sA[64 * 256];
  const int tid = threadIdx.x, lane = tid & 31, wv = tid >> 5;
  const int K = KT * 32;
  const int r0 = blockIdx.x * 64;
  for (int i = tid; i < (64 * K) / 8; i += 256) {
    int row = (i * 8) / K, col = (i * 8) % K;
    *(v8h*)&sA[row * K + col] = *(const v8h*)&A[(size_t)(r0 + row) * lda + col];
  }
  __syncthreads();
  for (int t = 0; t < 2; ++t) {
    int id = wv + 8 * t;               // 16 tiles: 4 mtiles x 4 ntiles
    int mt = id >> 2, ntl = id & 3;
    int ntg = blockIdx.y * 4 + ntl;
    v8f acc = {};
    for (int kt = 0; kt < KT; ++kt) {
      v16h a = fragA_rm(sA, mt * 16, K, kt * 32);
      v16h b = fragB_pk(Bp + ((size_t)ntg * KT + kt) * 512);
      acc = wmma16(a, b, acc);
    }
    int cn = ntg * 16 + (lane & 15);
    float bi = bias ? bias[cn] : 0.0f;
    float sc = cscale ? cscale[cn] : 1.0f;
    int mbase = r0 + mt * 16 + ((lane >> 4) << 3);
    #pragma unroll
    for (int r = 0; r < 8; ++r) {
      float v = (acc[r] + bi) * sc;
      if (out32) out32[(size_t)(mbase + r) * ldo + cn] = v;
      if (out16) out16[(size_t)(mbase + r) * ldo + cn] = (_Float16)v;
    }
  }
}

// ---------------- scores -> online softmax -> threshold adjacency ----------------
__global__ void __launch_bounds__(256)
scores_adj(const _Float16* __restrict__ qh, const _Float16* __restrict__ kh,
           const float* __restrict__ boP, const float* __restrict__ phiP,
           const float* __restrict__ maskE, unsigned char* __restrict__ adj,
           unsigned char* __restrict__ adjA, float* __restrict__ dinv,
           float* __restrict__ dinvA) {
  __shared__ __align__(16) _Float16 sQ[32 * cD];
  __shared__ float sStat[8][32][2];
  __shared__ float sMax[32], sInv[32];
  const int tid = threadIdx.x, lane = tid & 31, wv = tid >> 5;
  const int r0 = blockIdx.x * 32;
  const float bo = boP[0], phiv = phiP[0];
  for (int i = tid; i < (32 * cD) / 8; i += 256)
    *(v8h*)&sQ[i * 8] = *(const v8h*)&qh[(size_t)r0 * cD + i * 8];
  __syncthreads();

  float mx[2][8], sm[2][8];
  #pragma unroll
  for (int mt = 0; mt < 2; ++mt)
    #pragma unroll
    for (int r = 0; r < 8; ++r) { mx[mt][r] = -1e30f; sm[mt][r] = 0.0f; }

  // pass 1: row max/sum (online softmax over columns)
  for (int nt = wv; nt < 256; nt += 8) {
    v8f acc[2] = {{}, {}};
    for (int kt = 0; kt < 8; ++kt) {
      v16h b = fragB_rm(kh, nt * 16, cD, kt * 32);
      acc[0] = wmma16(fragA_rm(sQ, 0, cD, kt * 32), b, acc[0]);
      acc[1] = wmma16(fragA_rm(sQ, 16, cD, kt * 32), b, acc[1]);
    }
    #pragma unroll
    for (int mt = 0; mt < 2; ++mt)
      #pragma unroll
      for (int r = 0; r < 8; ++r) {
        float v = acc[mt][r] + bo;
        float nm = fmaxf(mx[mt][r], v);
        sm[mt][r] = sm[mt][r] * __expf(mx[mt][r] - nm) + __expf(v - nm);
        mx[mt][r] = nm;
      }
  }
  #pragma unroll
  for (int msk = 1; msk < 16; msk <<= 1)
    #pragma unroll
    for (int mt = 0; mt < 2; ++mt)
      #pragma unroll
      for (int r = 0; r < 8; ++r) {
        float om = __shfl_xor(mx[mt][r], msk, 32);
        float os = __shfl_xor(sm[mt][r], msk, 32);
        float nm = fmaxf(mx[mt][r], om);
        sm[mt][r] = sm[mt][r] * __expf(mx[mt][r] - nm) + os * __expf(om - nm);
        mx[mt][r] = nm;
      }
  if ((lane & 15) == 0) {
    #pragma unroll
    for (int mt = 0; mt < 2; ++mt)
      #pragma unroll
      for (int r = 0; r < 8; ++r) {
        int rl = mt * 16 + ((lane >> 4) << 3) + r;
        sStat[wv][rl][0] = mx[mt][r]; sStat[wv][rl][1] = sm[mt][r];
      }
  }
  __syncthreads();
  if (tid < 32) {
    float M = -1e30f, S = 0.0f;
    for (int w = 0; w < 8; ++w) {
      float m2 = sStat[w][tid][0], s2 = sStat[w][tid][1];
      float nm = fmaxf(M, m2);
      S = S * __expf(M - nm) + s2 * __expf(m2 - nm);
      M = nm;
    }
    sMax[tid] = M; sInv[tid] = 1.0f / S;
  }
  __syncthreads();

  // pass 2: recompute, threshold, write adjacency + degrees
  float dg[2][8] = {}, dga[2][8] = {};
  for (int nt = wv; nt < 256; nt += 8) {
    int col = nt * 16 + (lane & 15);
    v8f acc[2] = {{}, {}};
    for (int kt = 0; kt < 8; ++kt) {
      v16h b = fragB_rm(kh, nt * 16, cD, kt * 32);
      acc[0] = wmma16(fragA_rm(sQ, 0, cD, kt * 32), b, acc[0]);
      acc[1] = wmma16(fragA_rm(sQ, 16, cD, kt * 32), b, acc[1]);
    }
    #pragma unroll
    for (int mt = 0; mt < 2; ++mt)
      #pragma unroll
      for (int r = 0; r < 8; ++r) {
        int rl = mt * 16 + ((lane >> 4) << 3) + r;
        int gn = r0 + rl;
        float attn = __expf(acc[mt][r] + bo - sMax[rl]) * sInv[rl];
        int a = (attn >= phiv) && (col != gn);
        adj[(size_t)gn * cN + col] = (unsigned char)a;
        int aa = a && (maskE[(size_t)gn * cN + col] >= 0.1f);
        adjA[(size_t)gn * cN + col] = (unsigned char)aa;
        dg[mt][r] += (float)a; dga[mt][r] += (float)aa;
      }
  }
  #pragma unroll
  for (int msk = 1; msk < 16; msk <<= 1)
    #pragma unroll
    for (int mt = 0; mt < 2; ++mt)
      #pragma unroll
      for (int r = 0; r < 8; ++r) {
        dg[mt][r]  += __shfl_xor(dg[mt][r],  msk, 32);
        dga[mt][r] += __shfl_xor(dga[mt][r], msk, 32);
      }
  if ((lane & 15) == 0) {
    #pragma unroll
    for (int mt = 0; mt < 2; ++mt)
      #pragma unroll
      for (int r = 0; r < 8; ++r) {
        int rl = mt * 16 + ((lane >> 4) << 3) + r;
        sStat[wv][rl][0] = dg[mt][r]; sStat[wv][rl][1] = dga[mt][r];
      }
  }
  __syncthreads();
  if (tid < 32) {
    float d = 0, da = 0;
    for (int w = 0; w < 8; ++w) { d += sStat[w][tid][0]; da += sStat[w][tid][1]; }
    dinv[r0 + tid]  = rsqrtf(1.0f + d);
    dinvA[r0 + tid] = rsqrtf(1.0f + da);
  }
}

// ---------------- GCN: z = Dinv (A+I) Dinv XW + b  (A from u8, K = 4096) -------------
__global__ void __launch_bounds__(256)
gcn_apply(const unsigned char* __restrict__ adjM, const float* __restrict__ dv,
          const _Float16* __restrict__ xwP, const float* __restrict__ gb,
          float* __restrict__ z32, _Float16* __restrict__ z16) {
  __shared__ __align__(16) _Float16 sA[32 * 32];
  const int tid = threadIdx.x, lane = tid & 31, wv = tid >> 5;
  const int r0 = blockIdx.x * 32;
  v8f acc[2][2];
  {
    v8f zz = {};
    acc[0][0] = zz; acc[0][1] = zz; acc[1][0] = zz; acc[1][1] = zz;
  }
  for (int kt = 0; kt < cN / 32; ++kt) {
    __syncthreads();
    for (int e = tid; e < 1024; e += 256) {
      int lr = e >> 5, lc = e & 31;
      int gn = r0 + lr, gm = kt * 32 + lc;
      float a = (gm == gn) ? 1.0f : (float)adjM[(size_t)gn * cN + gm];
      sA[e] = (_Float16)(a * dv[gm]);
    }
    __syncthreads();
    #pragma unroll
    for (int mt = 0; mt < 2; ++mt) {
      v16h a = fragA_rm(sA, mt * 16, 32, 0);
      #pragma unroll
      for (int j = 0; j < 2; ++j) {
        int nt = wv * 2 + j;
        v16h b = fragB_pk(xwP + ((size_t)nt * (cN / 32) + kt) * 512);
        acc[mt][j] = wmma16(a, b, acc[mt][j]);
      }
    }
  }
  #pragma unroll
  for (int mt = 0; mt < 2; ++mt)
    #pragma unroll
    for (int j = 0; j < 2; ++j) {
      int cn = (wv * 2 + j) * 16 + (lane & 15);
      #pragma unroll
      for (int r = 0; r < 8; ++r) {
        int row = r0 + mt * 16 + ((lane >> 4) << 3) + r;
        float v = dv[row] * acc[mt][j][r] + gb[cn];
        z32[(size_t)row * cD + cn] = v;
        z16[(size_t)row * cD + cn] = (_Float16)v;
      }
    }
}

// ---------------- FFN + row L2-normalize ----------------
__global__ void __launch_bounds__(256)
ffn_norm(const _Float16* __restrict__ zin, const _Float16* __restrict__ fP,
         const float* __restrict__ fb, _Float16* __restrict__ znOut) {
  __shared__ __align__(16) _Float16 sA[32 * cD];
  __shared__ float sC[32 * cFFN];
  __shared__ float sRN[32];
  const int tid = threadIdx.x, lane = tid & 31, wv = tid >> 5;
  const int r0 = blockIdx.x * 32;
  for (int i = tid; i < (32 * cD) / 8; i += 256)
    *(v8h*)&sA[i * 8] = *(const v8h*)&zin[(size_t)r0 * cD + i * 8];
  __syncthreads();
  for (int t = 0; t < 2; ++t) {
    int id = wv + 8 * t;              // 2 mtiles x 8 ntiles
    int mt = id >> 3, nt = id & 7;
    v8f acc = {};
    for (int kt = 0; kt < 8; ++kt)
      acc = wmma16(fragA_rm(sA, mt * 16, cD, kt * 32),
                   fragB_pk(fP + ((size_t)nt * 8 + kt) * 512), acc);
    int cn = nt * 16 + (lane & 15);
    #pragma unroll
    for (int r = 0; r < 8; ++r)
      sC[(mt * 16 + ((lane >> 4) << 3) + r) * cFFN + cn] = acc[r] + fb[cn];
  }
  __syncthreads();
  if (tid < 32) {
    float s = 0.0f;
    for (int c = 0; c < cFFN; ++c) { float v = sC[tid * cFFN + c]; s += v * v; }
    sRN[tid] = 1.0f / fmaxf(sqrtf(s), 1e-12f);
  }
  __syncthreads();
  for (int e = tid; e < 32 * cFFN; e += 256)
    znOut[(size_t)(r0 + (e >> 7)) * cFFN + (e & 127)] = (_Float16)(sC[e] * sRN[e >> 7]);
}

// ---------------- clustering: soft assign + labels + column sums ----------------
__global__ void __launch_bounds__(256)
cluster_assign(const float* __restrict__ xs, const float* __restrict__ cen,
               float* __restrict__ qsoft, int* __restrict__ lab,
               float* __restrict__ fsum) {
  const int tid = threadIdx.x, lane = tid & 31, wv = tid >> 5;
  const int row = blockIdx.x * 8 + wv;
  float xv[8];
  #pragma unroll
  for (int i = 0; i < 8; ++i) xv[i] = xs[(size_t)row * cD + lane * 8 + i];
  float q[cK]; float qs = 0.0f;
  #pragma unroll
  for (int k = 0; k < cK; ++k) {
    float d = 0.0f;
    #pragma unroll
    for (int i = 0; i < 8; ++i) {
      float df = xv[i] - cen[k * cD + lane * 8 + i];
      d += df * df;
    }
    #pragma unroll
    for (int msk = 1; msk < 32; msk <<= 1) d += __shfl_xor(d, msk, 32);
    q[k] = 1.0f / (1.0f + d);
    qs += q[k];
  }
  int best = 0; float bq = -1.0f;
  #pragma unroll
  for (int k = 0; k < cK; ++k) {
    q[k] /= qs;
    if (q[k] > bq) { bq = q[k]; best = k; }
  }
  if (lane == 0) {
    lab[row] = best;
    #pragma unroll
    for (int k = 0; k < cK; ++k) {
      qsoft[(size_t)row * cK + k] = q[k];
      atomicAdd(&fsum[k], q[k]);
    }
  }
}

__global__ void kl_loss(const float* __restrict__ qsoft, const float* __restrict__ fsum,
                        float* __restrict__ klsum) {
  int n = blockIdx.x * 256 + threadIdx.x;
  float p[cK]; float ps = 0.0f;
  #pragma unroll
  for (int k = 0; k < cK; ++k) {
    float q = qsoft[(size_t)n * cK + k];
    float pv = q * q / fsum[k];
    p[k] = pv; ps += pv;
  }
  float kl = 0.0f;
  #pragma unroll
  for (int k = 0; k < cK; ++k) {
    float pv = p[k] / ps;
    float q = qsoft[(size_t)n * cK + k];
    kl += pv * logf(pv / (q + 1e-6f));
  }
  atomicAdd(klsum, kl * (1.0f / cN));
}

// ---------------- contrastive loss (two fused NxN exp-GEMMs, K=128) ----------------
__global__ void __launch_bounds__(256)
cl_loss(const _Float16* __restrict__ zn, const _Float16* __restrict__ zan,
        const unsigned char* __restrict__ adjM, const int* __restrict__ lab,
        float* __restrict__ clsum) {
  __shared__ __align__(16) _Float16 sZ[32 * cFFN];
  __shared__ int sLab[32];
  __shared__ float sRed[8][32][6];
  const int tid = threadIdx.x, lane = tid & 31, wv = tid >> 5;
  const int r0 = blockIdx.x * 32;
  for (int i = tid; i < (32 * cFFN) / 8; i += 256)
    *(v8h*)&sZ[i * 8] = *(const v8h*)&zn[(size_t)r0 * cFFN + i * 8];
  if (tid < 32) sLab[tid] = lab[r0 + tid];
  __syncthreads();

  float si[2][8] = {}, sx[2][8] = {}, ps[2][8] = {}, ct[2][8] = {}, di[2][8] = {}, dx[2][8] = {};
  for (int nt = wv; nt < 256; nt += 8) {
    int col = nt * 16 + (lane & 15);
    int lm = lab[col];
    v8f aI[2] = {{}, {}}, aX[2] = {{}, {}};
    for (int kt = 0; kt < 4; ++kt) {
      v16h bI = fragB_rm(zn,  nt * 16, cFFN, kt * 32);
      v16h bX = fragB_rm(zan, nt * 16, cFFN, kt * 32);
      v16h a0 = fragA_rm(sZ, 0,  cFFN, kt * 32);
      v16h a1 = fragA_rm(sZ, 16, cFFN, kt * 32);
      aI[0] = wmma16(a0, bI, aI[0]); aI[1] = wmma16(a1, bI, aI[1]);
      aX[0] = wmma16(a0, bX, aX[0]); aX[1] = wmma16(a1, bX, aX[1]);
    }
    #pragma unroll
    for (int mt = 0; mt < 2; ++mt)
      #pragma unroll
      for (int r = 0; r < 8; ++r) {
        int rl = mt * 16 + ((lane >> 4) << 3) + r;
        int gn = r0 + rl;
        float ei = __expf(aI[mt][r] * 2.0f);   // 1/TAU = 2
        float ex = __expf(aX[mt][r] * 2.0f);
        int ab = (int)adjM[(size_t)gn * cN + col];
        int cn2 = (lm == sLab[rl]) && (col != gn) && (!ab);
        si[mt][r] += ei; sx[mt][r] += ex;
        if (col == gn) { di[mt][r] += ei; dx[mt][r] += ex; }
        if (ab | cn2) { ps[mt][r] += ei; ct[mt][r] += 1.0f; }
      }
  }
  #pragma unroll
  for (int msk = 1; msk < 16; msk <<= 1)
    #pragma unroll
    for (int mt = 0; mt < 2; ++mt)
      #pragma unroll
      for (int r = 0; r < 8; ++r) {
        si[mt][r] += __shfl_xor(si[mt][r], msk, 32);
        sx[mt][r] += __shfl_xor(sx[mt][r], msk, 32);
        ps[mt][r] += __shfl_xor(ps[mt][r], msk, 32);
        ct[mt][r] += __shfl_xor(ct[mt][r], msk, 32);
        di[mt][r] += __shfl_xor(di[mt][r], msk, 32);
        dx[mt][r] += __shfl_xor(dx[mt][r], msk, 32);
      }
  if ((lane & 15) == 0) {
    #pragma unroll
    for (int mt = 0; mt < 2; ++mt)
      #pragma unroll
      for (int r = 0; r < 8; ++r) {
        int rl = mt * 16 + ((lane >> 4) << 3) + r;
        sRed[wv][rl][0] = si[mt][r]; sRed[wv][rl][1] = sx[mt][r];
        sRed[wv][rl][2] = ps[mt][r]; sRed[wv][rl][3] = ct[mt][r];
        sRed[wv][rl][4] = di[mt][r]; sRed[wv][rl][5] = dx[mt][r];
      }
  }
  __syncthreads();
  if (tid < 32) {
    float Si = 0, Sx = 0, P = 0, C = 0, Di = 0, Dx = 0;
    for (int w = 0; w < 8; ++w) {
      Si += sRed[w][tid][0]; Sx += sRed[w][tid][1];
      P  += sRed[w][tid][2]; C  += sRed[w][tid][3];
      Di += sRed[w][tid][4]; Dx += sRed[w][tid][5];
    }
    float positive = Dx + P;
    float denom = Si + Sx - Di;
    float cnt = C + 1.0f;
    atomicAdd(clsum, -logf(positive / denom) / cnt);
  }
}

// ---------------- head + loss finalize ----------------
__global__ void predict_k(const float* __restrict__ z32, const float* __restrict__ pw,
                          const float* __restrict__ pb, float* __restrict__ outp) {
  int g = blockIdx.x * 256 + threadIdx.x;
  if (g >= cN * 2) return;
  int n = g >> 1, j = g & 1;
  float s = pb[j];
  for (int c = 0; c < cD; ++c) s += z32[(size_t)n * cD + c] * pw[c * 2 + j];
  outp[g] = s;
}

__global__ void finalize_loss(const float* kl, const float* cl, float* outp) {
  if (threadIdx.x == 0) outp[cN * 2] = kl[0] + cl[0];   // lambda_kl = lambda_cl = 1
}

// ======================= launch =======================
extern "C" void kernel_launch(void* const* d_in, const int* in_sizes, int n_in,
                              void* d_out, int out_size, void* d_ws, size_t ws_size,
                              hipStream_t stream) {
  (void)in_sizes; (void)n_in; (void)out_size; (void)ws_size;
  const float* x       = (const float*)d_in[0];
  const float* x_demo  = (const float*)d_in[1];
  const int*   slen    = (const int*)  d_in[2];
  const float* mask_e  = (const float*)d_in[3];
  const float* w_ih    = (const float*)d_in[4];
  const float* w_hh    = (const float*)d_in[5];
  const float* b_ih    = (const float*)d_in[6];
  const float* b_hh    = (const float*)d_in[7];
  const float* h0      = (const float*)d_in[8];
  const float* wq      = (const float*)d_in[9];
  const float* bq      = (const float*)d_in[10];
  const float* wk      = (const float*)d_in[11];
  const float* bk      = (const float*)d_in[12];
  const float* wo      = (const float*)d_in[13];
  const float* bo      = (const float*)d_in[14];
  const float* phi     = (const float*)d_in[15];
  const float* centers = (const float*)d_in[16];
  const float* gcn_w   = (const float*)d_in[17];
  const float* gcn_b   = (const float*)d_in[18];
  const float* ffn_w   = (const float*)d_in[19];
  const float* ffn_b   = (const float*)d_in[20];
  const float* pre_w   = (const float*)d_in[21];
  const float* pre_b   = (const float*)d_in[22];
  float* out = (float*)d_out;
  char*  ws  = (char*)d_ws;

  size_t off = 0;
  auto carve = [&](size_t bytes) { size_t p = off; off += (bytes + 255) & ~(size_t)255; return p; };

  _Float16* xh    = (_Float16*)(ws + carve((size_t)cT * cN * cFP * 2));
  _Float16* WihP  = (_Float16*)(ws + carve((size_t)3 * 48 * 512 * 2));
  _Float16* WhhP  = (_Float16*)(ws + carve((size_t)8 * 48 * 512 * 2));
  _Float16* WqP   = (_Float16*)(ws + carve((size_t)8 * 16 * 512 * 2));
  _Float16* WkP   = (_Float16*)(ws + carve((size_t)8 * 16 * 512 * 2));
  _Float16* GcnP  = (_Float16*)(ws + carve((size_t)8 * 16 * 512 * 2));
  _Float16* FfnP  = (_Float16*)(ws + carve((size_t)8 * 8 * 512 * 2));
  float*    xs32  = (float*)   (ws + carve((size_t)cN * cD * 4));
  _Float16* xs16  = (_Float16*)(ws + carve((size_t)cN * cD * 2));
  _Float16* qhB   = (_Float16*)(ws + carve((size_t)cN * cD * 2));
  _Float16* khB   = (_Float16*)(ws + carve((size_t)cN * cD * 2));
  float*    cscale= (float*)   (ws + carve(256 * 4));
  unsigned char* adj  = (unsigned char*)(ws + carve((size_t)cN * cN));
  unsigned char* adjA = (unsigned char*)(ws + carve((size_t)cN * cN));
  float*    dinv  = (float*)   (ws + carve(cN * 4));
  float*    dinvA = (float*)   (ws + carve(cN * 4));
  _Float16* xw16  = (_Float16*)(ws + carve((size_t)cN * cD * 2));
  _Float16* xwP   = (_Float16*)(ws + carve((size_t)(cN / 32) * 16 * 512 * 2));
  float*    z32   = (float*)   (ws + carve((size_t)cN * cD * 4));
  _Float16* z16   = (_Float16*)(ws + carve((size_t)cN * cD * 2));
  float*    za32  = (float*)   (ws + carve((size_t)cN * cD * 4));
  _Float16* za16  = (_Float16*)(ws + carve((size_t)cN * cD * 2));
  _Float16* zn16  = (_Float16*)(ws + carve((size_t)cN * cFFN * 2));
  _Float16* zan16 = (_Float16*)(ws + carve((size_t)cN * cFFN * 2));
  float*    qsoft = (float*)   (ws + carve((size_t)cN * cK * 4));
  int*      label = (int*)     (ws + carve((size_t)cN * 4));
  float*    fsum  = (float*)   (ws + carve(cK * 4));
  float*    klsum = (float*)   (ws + carve(4));
  float*    clsum = (float*)   (ws + carve(4));

  init_zero<<<1, 64, 0, stream>>>(fsum, klsum, clsum);

  pack_w<<<3 * 48 * 2, 256, 0, stream>>>(w_ih, cF, 756, 1, WihP, 3);
  pack_w<<<8 * 48 * 2, 256, 0, stream>>>(w_hh, cH, 756, 1, WhhP, 8);
  pack_w<<<8 * 16 * 2, 256, 0, stream>>>(wq, cD, cD, 1, WqP, 8);
  pack_w<<<8 * 16 * 2, 256, 0, stream>>>(wk, cD, cD, 1, WkP, 8);
  pack_w<<<8 * 16 * 2, 256, 0, stream>>>(gcn_w, cD, cD, 0, GcnP, 8);
  pack_w<<<8 * 8 * 2, 256, 0, stream>>>(ffn_w, cD, cFFN, 0, FfnP, 8);

  convert_x<<<(cT * cN * cFP) / 256, 256, 0, stream>>>(x, xh);
  fill_demo<<<cN / 256, 256, 0, stream>>>(x_demo, xs32, xs16);
  build_cscale<<<1, 256, 0, stream>>>(wo, cscale);

  gru_kernel<<<cN / 16, 256, 0, stream>>>(xh, WihP, WhhP, b_ih, b_hh, h0, slen, xs32, xs16);

  gemm16<<<dim3(cN / 64, 4), 256, 0, stream>>>(xs16, cD, 8, WqP, bq, cscale, qhB, nullptr, cD);
  gemm16<<<dim3(cN / 64, 4), 256, 0, stream>>>(xs16, cD, 8, WkP, bk, nullptr, khB, nullptr, cD);
  gemm16<<<dim3(cN / 64, 4), 256, 0, stream>>>(xs16, cD, 8, GcnP, nullptr, nullptr, xw16, nullptr, cD);
  pack_b16<<<(cN / 32) * 16 * 2, 256, 0, stream>>>(xw16, cD, cN / 32, xwP);

  scores_adj<<<cN / 32, 256, 0, stream>>>(qhB, khB, bo, phi, mask_e, adj, adjA, dinv, dinvA);

  gcn_apply<<<cN / 32, 256, 0, stream>>>(adj, dinv, xwP, gcn_b, z32, z16);
  gcn_apply<<<cN / 32, 256, 0, stream>>>(adjA, dinvA, xwP, gcn_b, za32, za16);

  ffn_norm<<<cN / 32, 256, 0, stream>>>(z16, FfnP, ffn_b, zn16);
  ffn_norm<<<cN / 32, 256, 0, stream>>>(za16, FfnP, ffn_b, zan16);

  cluster_assign<<<cN / 8, 256, 0, stream>>>(xs32, centers, qsoft, label, fsum);
  kl_loss<<<cN / 256, 256, 0, stream>>>(qsoft, fsum, klsum);

  cl_loss<<<cN / 32, 256, 0, stream>>>(zn16, zan16, adj, label, clsum);

  predict_k<<<(cN * 2) / 256, 256, 0, stream>>>(z32, pre_w, pre_b, out);
  finalize_loss<<<1, 32, 0, stream>>>(klsum, clsum, out);
}
